// PredictiveLayer_70394513981695
// MI455X (gfx1250) — matrix operand order repaired and checked by reading, not compiled, not run on hardware
//
#include <hip/hip_runtime.h>
#include <math.h>

// ---------------------------------------------------------------------------
// Types for CDNA5 WMMA (wave32, gfx1250)
// ---------------------------------------------------------------------------
typedef __attribute__((ext_vector_type(16))) __bf16 v16bf;
typedef __attribute__((ext_vector_type(8)))  float  v8f;
typedef __attribute__((ext_vector_type(4)))  int    v4i;

#define H_DIM   576
#define I_DIM   1536
#define BD_DIM  72
#define R_DIM   16
#define S_DIM   2048

// CDNA5 async global->LDS copy path (ASYNCcnt-tracked, bypasses VGPRs)
#if __has_builtin(__builtin_amdgcn_global_load_async_to_lds_b128)
#define HAVE_ASYNC_LDS 1
#endif

__device__ __forceinline__ void cp16_to_lds(const unsigned short* g, unsigned short* l) {
#ifdef HAVE_ASYNC_LDS
    __builtin_amdgcn_global_load_async_to_lds_b128(
        (__attribute__((address_space(1))) v4i*)g,
        (__attribute__((address_space(3))) v4i*)l, 0, 0);
#else
    *(uint4*)l = *(const uint4*)g;
#endif
}

__device__ __forceinline__ void wait_async_copies() {
#ifdef HAVE_ASYNC_LDS
#if __has_builtin(__builtin_amdgcn_s_wait_asynccnt)
    __builtin_amdgcn_s_wait_asynccnt(0);
#else
    asm volatile("s_wait_asynccnt 0x0" ::: "memory");
#endif
#endif
}

// ---------------------------------------------------------------------------
// bf16 helpers (round-to-nearest-even), hi/lo split for bf16x3 GEMM
// ---------------------------------------------------------------------------
__device__ __forceinline__ unsigned short bf16_rne(float f) {
    unsigned int u = __float_as_uint(f);
    unsigned int r = u + 0x7FFFu + ((u >> 16) & 1u);
    return (unsigned short)(r >> 16);
}
__device__ __forceinline__ float bf16_to_f(unsigned short h) {
    return __uint_as_float(((unsigned int)h) << 16);
}

// ---------------------------------------------------------------------------
// Elementwise / prep kernels
// ---------------------------------------------------------------------------
__global__ void prec_kernel(const float* __restrict__ logp, float* __restrict__ prec, int n) {
    int i = blockIdx.x * blockDim.x + threadIdx.x;
    if (i < n) prec[i] = expf(logp[i]);
}

__global__ void cvt_hl_kernel(const float* __restrict__ in,
                              unsigned short* __restrict__ hi,
                              unsigned short* __restrict__ lo, size_t n) {
    size_t i = (size_t)blockIdx.x * blockDim.x + threadIdx.x;
    if (i >= n) return;
    float x = in[i];
    unsigned short h = bf16_rne(x);
    hi[i] = h;
    lo[i] = bf16_rne(x - bf16_to_f(h));
}

// RMSNorm over H, emit ns (normed*weight) and raw state as bf16 hi/lo pairs.
__global__ void __launch_bounds__(256)
rmsnorm_kernel(const float* __restrict__ state, const float* __restrict__ w,
               unsigned short* __restrict__ nsH, unsigned short* __restrict__ nsL,
               unsigned short* __restrict__ stH, unsigned short* __restrict__ stL) {
    __shared__ float red[256];
    const int row = blockIdx.x;
    const float* x = state + (size_t)row * H_DIM;
    float s = 0.0f;
    for (int i = threadIdx.x; i < H_DIM; i += 256) { float v = x[i]; s += v * v; }
    red[threadIdx.x] = s;
    __syncthreads();
    for (int o = 128; o > 0; o >>= 1) {
        if (threadIdx.x < o) red[threadIdx.x] += red[threadIdx.x + o];
        __syncthreads();
    }
    const float r = rsqrtf(red[0] / (float)H_DIM + 1e-6f);
    for (int i = threadIdx.x; i < H_DIM; i += 256) {
        float v = x[i];
        float ns = v * r * w[i];
        size_t idx = (size_t)row * H_DIM + i;
        unsigned short nh = bf16_rne(ns);
        nsH[idx] = nh; nsL[idx] = bf16_rne(ns - bf16_to_f(nh));
        unsigned short sh = bf16_rne(v);
        stH[idx] = sh; stL[idx] = bf16_rne(v - bf16_to_f(sh));
    }
}

__global__ void errup_kernel(const float* __restrict__ state, const float* __restrict__ upred,
                             const float* __restrict__ uprec, float* __restrict__ out, size_t n) {
    size_t i = (size_t)blockIdx.x * blockDim.x + threadIdx.x;
    if (i >= n) return;
    int h = (int)(i % H_DIM);
    out[i] = uprec[h] * (state[i] - upred[i]);
}

// ---------------------------------------------------------------------------
// bf16x3 WMMA GEMM: C(f32) = A(hi+lo) x B(hi+lo), fused epilogues.
//   BM=128, BN=64, BK=32; 256 threads = 8 waves; wave w -> 16-row strip,
//   4 N-tiles of 16x16 each (3 WMMAs per tile: AhBh + AhBl + AlBh).
//   Double-buffered LDS fed by GLOBAL_LOAD_ASYNC_TO_LDS_B128; two barriers
//   per K-step, async copies overlapped with WMMA compute.
// Fragment layouts follow CDNA5 ISA §7.12.2:
//   A 16x32 bf16: lane<16 row m=lane, K {0..7,16..23}; lane>=16 K {8..15,24..31}
//   B 32x16 bf16: lane holds column N=lane%16, same K sets as A
//   C/D 16x16 f32: VGPR r -> M = (lane/16)*8 + r, N = lane%16
// EPI: 1 = exact GELU -> bf16 hi/lo store (activation for GEMM2)
//      2 = prediction f32 + error_down = prec[col]*(lower - pred)
//      3 = inhibited = state - 0.1*relu(acc)
// ---------------------------------------------------------------------------
#define LDA 40   // 32 + 8 pad (ushorts)
#define LDB 72   // 64 + 8 pad (ushorts)

template <int EPI>
__global__ void __launch_bounds__(256)
gemm_bf16x3(const unsigned short* __restrict__ AH, const unsigned short* __restrict__ AL,
            const unsigned short* __restrict__ BHp, const unsigned short* __restrict__ BLp,
            int N, int K,
            float* __restrict__ out0, float* __restrict__ out1,
            const float* __restrict__ aux0, const float* __restrict__ aux1,
            unsigned short* __restrict__ obfH, unsigned short* __restrict__ obfL) {
    __shared__ unsigned short sAH[2][128 * LDA];
    __shared__ unsigned short sAL[2][128 * LDA];
    __shared__ unsigned short sBH[2][32 * LDB];
    __shared__ unsigned short sBL[2][32 * LDB];

    const int tid  = threadIdx.x;
    const int lane = tid & 31;
    const int wave = tid >> 5;
    const int mBase = blockIdx.y * 128;
    const int nBase = blockIdx.x * 64;

    // per-thread copy assignment (constant across K-steps)
    const int arow = tid >> 1, aseg = tid & 1;           // A: 128 rows x 2 segments
    const int brow = tid >> 3, bseg = tid & 7;           // B: 32 rows x 8 segments
    const size_t aoff = (size_t)(mBase + arow) * K + aseg * 16;
    const int    al   = arow * LDA + aseg * 16;
    const int    bl   = brow * LDB + bseg * 8;

    auto issue_tile = [&](int buf, int k0) {
        const size_t ga = aoff + k0;
        cp16_to_lds(AH + ga,     &sAH[buf][al]);
        cp16_to_lds(AH + ga + 8, &sAH[buf][al + 8]);
        cp16_to_lds(AL + ga,     &sAL[buf][al]);
        cp16_to_lds(AL + ga + 8, &sAL[buf][al + 8]);
        const size_t gb = (size_t)(k0 + brow) * N + nBase + bseg * 8;
        cp16_to_lds(BHp + gb, &sBH[buf][bl]);
        cp16_to_lds(BLp + gb, &sBL[buf][bl]);
    };

    v8f acc[4] = {};
    union Frag { v16bf v; uint4 q[2]; };

    const int nK = K / 32;
    issue_tile(0, 0);

    for (int kt = 0; kt < nK; ++kt) {
        const int cur = kt & 1;
        wait_async_copies();     // my async copies into `cur` are done
        __syncthreads();         // everyone's copies done; prev buffer free
        if (kt + 1 < nK) issue_tile(cur ^ 1, (kt + 1) * 32);  // overlap with compute

        Frag aH, aL;
        const int am = (wave * 16 + (lane & 15)) * LDA;
        const int ah = (lane >> 4) * 8;
        aH.q[0] = *(const uint4*)&sAH[cur][am + ah];
        aH.q[1] = *(const uint4*)&sAH[cur][am + 16 + ah];
        aL.q[0] = *(const uint4*)&sAL[cur][am + ah];
        aL.q[1] = *(const uint4*)&sAL[cur][am + 16 + ah];

#pragma unroll
        for (int nt = 0; nt < 4; ++nt) {
            Frag bH, bL;
            const int bo = nt * 16 + (lane & 15);
            const int bk = (lane >> 4) * 8;
#pragma unroll
            for (int s = 0; s < 16; ++s) {
                const int k = (s < 8) ? (bk + s) : (16 + bk + s - 8);
                bH.v[s] = __builtin_bit_cast(__bf16, sBH[cur][k * LDB + bo]);
                bL.v[s] = __builtin_bit_cast(__bf16, sBL[cur][k * LDB + bo]);
            }
            acc[nt] = __builtin_amdgcn_wmma_f32_16x16x32_bf16(
                false, aH.v, false, bH.v, (short)0, acc[nt], false, false);
            acc[nt] = __builtin_amdgcn_wmma_f32_16x16x32_bf16(
                false, aH.v, false, bL.v, (short)0, acc[nt], false, false);
            acc[nt] = __builtin_amdgcn_wmma_f32_16x16x32_bf16(
                false, aL.v, false, bH.v, (short)0, acc[nt], false, false);
        }
        __syncthreads();         // all waves done reading `cur` before it is refilled
    }

    const int n_loc = lane & 15;
    const int half  = lane >> 4;
#pragma unroll
    for (int nt = 0; nt < 4; ++nt) {
        const int col = nBase + nt * 16 + n_loc;
#pragma unroll
        for (int r = 0; r < 8; ++r) {
            const int rowg = mBase + wave * 16 + half * 8 + r;
            const float v = acc[nt][r];
            const size_t idx = (size_t)rowg * N + col;
            if (EPI == 1) {
                float g = 0.5f * v * (1.0f + erff(v * 0.70710678118654752f));
                unsigned short h = bf16_rne(g);
                obfH[idx] = h;
                obfL[idx] = bf16_rne(g - bf16_to_f(h));
            } else if (EPI == 2) {
                out0[idx] = v;                               // prediction
                out1[idx] = aux0[col] * (aux1[idx] - v);     // precision * (lower - pred)
            } else {
                out0[idx] = aux1[idx] - 0.1f * fmaxf(v, 0.0f); // inhibited
            }
        }
    }
}

// ---------------------------------------------------------------------------
// Exact global quantile via 3-round radix select on |x| float bit patterns
// (non-negative floats compare as their uint bit patterns).
// ---------------------------------------------------------------------------
__global__ void zero_hist(unsigned int* __restrict__ hist, int n) {
    int i = blockIdx.x * blockDim.x + threadIdx.x;
    if (i < n) hist[i] = 0u;
}

__global__ void __launch_bounds__(256)
hist_kernel(const float* __restrict__ data, size_t n, const uint2* __restrict__ ctl,
            unsigned int condShift, unsigned int binShift, unsigned int binMask,
            unsigned int* __restrict__ hist) {
    __shared__ unsigned int sh[2048];
    for (int i = threadIdx.x; i < 2048; i += 256) sh[i] = 0u;
    __syncthreads();
    const unsigned int prefix = ctl->x;
    const size_t stride = (size_t)gridDim.x * blockDim.x;
    for (size_t i = (size_t)blockIdx.x * blockDim.x + threadIdx.x; i < n; i += stride) {
        unsigned int bits = __float_as_uint(fabsf(data[i]));
        unsigned int hb = (condShift < 32u) ? (bits >> condShift) : 0u;
        if (hb == prefix) atomicAdd(&sh[(bits >> binShift) & binMask], 1u);
    }
    __syncthreads();
    for (int i = threadIdx.x; i < 2048; i += 256)
        if (sh[i]) atomicAdd(&hist[i], sh[i]);
}

__global__ void scan_kernel(const unsigned int* __restrict__ hist, uint2* __restrict__ ctl,
                            unsigned int nbins, unsigned int binBits, int isFinal,
                            float* __restrict__ result) {
    unsigned long long c = 0;
    unsigned long long rank = ctl->y;
    unsigned int b = nbins - 1;
    for (unsigned int i = 0; i < nbins; ++i) {
        unsigned long long h = hist[i];
        if (rank < c + h) { b = i; rank -= c; break; }
        c += h;
    }
    unsigned int prefix = (ctl->x << binBits) | b;
    ctl->x = prefix;
    ctl->y = (unsigned int)rank;
    if (isFinal) *result = __uint_as_float(prefix);
}

__global__ void qinit_kernel(uint2* __restrict__ ctls, unsigned int rank0) {
    for (int c = 0; c < 4; ++c) { ctls[c].x = 0u; ctls[c].y = rank0 + (unsigned int)(c & 1); }
}

__global__ void thr_kernel(const float* __restrict__ res, float frac, float* __restrict__ thr) {
    thr[0] = res[0] + frac * (res[1] - res[0]);
    thr[1] = res[2] + frac * (res[3] - res[2]);
}

__global__ void sparse_kernel(const float* __restrict__ err, const float* __restrict__ thr,
                              float* __restrict__ sparse, float* __restrict__ mask, size_t n) {
    size_t i = (size_t)blockIdx.x * blockDim.x + threadIdx.x;
    if (i >= n) return;
    float t = *thr;
    float e = err[i];
    float m = (fabsf(e) > t) ? 1.0f : 0.0f;
    sparse[i] = e * m;
    mask[i]   = m;
}

// ---------------------------------------------------------------------------
// Boundary encoding: region mean / unbiased var, then small projection.
// ---------------------------------------------------------------------------
__global__ void region_stats_kernel(const float* __restrict__ state,
                                    float* __restrict__ mean, float* __restrict__ var,
                                    int Bb) {
    const int total = Bb * R_DIM * H_DIM;
    int idx = blockIdx.x * blockDim.x + threadIdx.x;
    if (idx >= total) return;
    const int h = idx % H_DIM;
    const int r = (idx / H_DIM) % R_DIM;
    const int b = idx / (H_DIM * R_DIM);
    const int rs = S_DIM / R_DIM; // 128
    const float* p = state + ((size_t)(b * S_DIM + r * rs)) * H_DIM + h;
    float s = 0.0f, s2 = 0.0f;
    for (int j = 0; j < rs; ++j) { float v = p[(size_t)j * H_DIM]; s += v; s2 += v * v; }
    float m = s / (float)rs;
    mean[idx] = m;
    var[idx] = (s2 - (float)rs * m * m) / (float)(rs - 1);
}

__global__ void boundary_kernel(const float* __restrict__ mean, const float* __restrict__ var,
                                const float* __restrict__ Wm, const float* __restrict__ Wv,
                                float* __restrict__ outb, int rows) {
    int idx = blockIdx.x * blockDim.x + threadIdx.x;
    if (idx >= rows * BD_DIM) return;
    const int rrow = idx / BD_DIM, j = idx % BD_DIM;
    const float* mrow = mean + (size_t)rrow * H_DIM;
    const float* vrow = var + (size_t)rrow * H_DIM;
    float acc = 0.0f;
    for (int h = 0; h < H_DIM; ++h)
        acc = fmaf(mrow[h], Wm[h * BD_DIM + j], fmaf(vrow[h], Wv[h * BD_DIM + j], acc));
    outb[idx] = acc;
}

// ---------------------------------------------------------------------------
// Host orchestration
// ---------------------------------------------------------------------------
extern "C" void kernel_launch(void* const* d_in, const int* in_sizes, int n_in,
                              void* d_out, int out_size, void* d_ws, size_t ws_size,
                              hipStream_t stream) {
    (void)n_in; (void)out_size; (void)ws_size;
    const float* state = (const float*)d_in[0];
    const float* lower = (const float*)d_in[1];
    const float* upred = (const float*)d_in[2];
    const float* uprec = (const float*)d_in[3];
    const float* logp  = (const float*)d_in[4];
    const float* normw = (const float*)d_in[5];
    const float* W1    = (const float*)d_in[6];
    const float* W2    = (const float*)d_in[7];
    const float* Wm    = (const float*)d_in[8];
    const float* Wv    = (const float*)d_in[9];
    const float* Wi    = (const float*)d_in[10];

    const size_t NE = (size_t)in_sizes[0];      // B*S*H = 4718592
    const int M  = (int)(NE / H_DIM);           // 8192
    const int Bb = M / S_DIM;                   // 4

    float* out    = (float*)d_out;
    float* o_pred = out;
    float* o_ed   = out + 1 * NE;
    float* o_sd   = out + 2 * NE;
    float* o_md   = out + 3 * NE;
    float* o_eu   = out + 4 * NE;
    float* o_su   = out + 5 * NE;
    float* o_mu   = out + 6 * NE;
    float* o_bd   = out + 7 * NE;
    float* o_inh  = out + 7 * NE + (size_t)Bb * R_DIM * BD_DIM;

    // workspace carve-out (256B aligned)
    char* ws = (char*)d_ws;
    size_t off = 0;
    auto alloc = [&](size_t bytes) -> void* {
        void* p = ws + off;
        off = (off + bytes + 255) & ~(size_t)255;
        return p;
    };
    const size_t szNSH = (size_t)M * H_DIM * 2;
    const size_t szACT = (size_t)M * I_DIM * 2;
    unsigned short* nsH = (unsigned short*)alloc(szNSH);
    unsigned short* nsL = (unsigned short*)alloc(szNSH);
    unsigned short* stH = (unsigned short*)alloc(szNSH);
    unsigned short* stL = (unsigned short*)alloc(szNSH);
    unsigned short* acH = (unsigned short*)alloc(szACT);
    unsigned short* acL = (unsigned short*)alloc(szACT);
    unsigned short* W1H = (unsigned short*)alloc((size_t)H_DIM * I_DIM * 2);
    unsigned short* W1L = (unsigned short*)alloc((size_t)H_DIM * I_DIM * 2);
    unsigned short* W2H = (unsigned short*)alloc((size_t)I_DIM * H_DIM * 2);
    unsigned short* W2L = (unsigned short*)alloc((size_t)I_DIM * H_DIM * 2);
    unsigned short* WiH = (unsigned short*)alloc((size_t)H_DIM * H_DIM * 2);
    unsigned short* WiL = (unsigned short*)alloc((size_t)H_DIM * H_DIM * 2);
    float* prec  = (float*)alloc(H_DIM * 4);
    float* rmean = (float*)alloc((size_t)Bb * R_DIM * H_DIM * 4);
    float* rvar  = (float*)alloc((size_t)Bb * R_DIM * H_DIM * 4);
    unsigned int* hist = (unsigned int*)alloc(2048 * 4);
    uint2* ctls  = (uint2*)alloc(4 * sizeof(uint2));
    float* res   = (float*)alloc(4 * 4);
    float* thr   = (float*)alloc(2 * 4);

    // --- prep ---
    prec_kernel<<<(H_DIM + 255) / 256, 256, 0, stream>>>(logp, prec, H_DIM);
    cvt_hl_kernel<<<((size_t)H_DIM * I_DIM + 255) / 256, 256, 0, stream>>>(W1, W1H, W1L, (size_t)H_DIM * I_DIM);
    cvt_hl_kernel<<<((size_t)I_DIM * H_DIM + 255) / 256, 256, 0, stream>>>(W2, W2H, W2L, (size_t)I_DIM * H_DIM);
    cvt_hl_kernel<<<((size_t)H_DIM * H_DIM + 255) / 256, 256, 0, stream>>>(Wi, WiH, WiL, (size_t)H_DIM * H_DIM);
    rmsnorm_kernel<<<M, 256, 0, stream>>>(state, normw, nsH, nsL, stH, stL);

    // --- GEMM chain (bf16x3 WMMA) ---
    // act = gelu(ns @ W1)                 : [M,576] x [576,1536]
    gemm_bf16x3<1><<<dim3(I_DIM / 64, M / 128), 256, 0, stream>>>(
        nsH, nsL, W1H, W1L, I_DIM, H_DIM, nullptr, nullptr, nullptr, nullptr, acH, acL);
    // pred = act @ W2 ; error_down fused : [M,1536] x [1536,576]
    gemm_bf16x3<2><<<dim3(H_DIM / 64, M / 128), 256, 0, stream>>>(
        acH, acL, W2H, W2L, H_DIM, I_DIM, o_pred, o_ed, prec, lower, nullptr, nullptr);
    // inhibited = state - 0.1*relu(state @ Wi) : [M,576] x [576,576]
    gemm_bf16x3<3><<<dim3(H_DIM / 64, M / 128), 256, 0, stream>>>(
        stH, stL, WiH, WiL, H_DIM, H_DIM, o_inh, nullptr, nullptr, state, nullptr, nullptr);

    // --- top-down error ---
    errup_kernel<<<(NE + 255) / 256, 256, 0, stream>>>(state, upred, uprec, o_eu, NE);

    // --- exact 0.97-quantile thresholds (radix select, ranks r0 and r0+1) ---
    const double pos = 0.97 * (double)(NE - 1);
    const unsigned long long r0 = (unsigned long long)pos;
    const float frac = (float)(pos - (double)r0);
    qinit_kernel<<<1, 1, 0, stream>>>(ctls, (unsigned int)r0);
    const float* qdata[2] = { o_ed, o_eu };
    struct Round { unsigned int condShift, binShift, binMask, nbins, binBits; };
    const Round rounds[3] = {
        {32u, 21u, 0x7FFu, 2048u, 11u},
        {21u, 10u, 0x7FFu, 2048u, 11u},
        {10u,  0u, 0x3FFu, 1024u, 10u},
    };
    for (int c = 0; c < 4; ++c) {
        const float* data = qdata[c >> 1];
        uint2* ctl = ctls + c;
        for (int rd = 0; rd < 3; ++rd) {
            zero_hist<<<8, 256, 0, stream>>>(hist, 2048);
            hist_kernel<<<1024, 256, 0, stream>>>(data, NE, ctl,
                rounds[rd].condShift, rounds[rd].binShift, rounds[rd].binMask, hist);
            scan_kernel<<<1, 1, 0, stream>>>(hist, ctl, rounds[rd].nbins,
                rounds[rd].binBits, rd == 2, res + c);
        }
    }
    thr_kernel<<<1, 1, 0, stream>>>(res, frac, thr);
    sparse_kernel<<<(NE + 255) / 256, 256, 0, stream>>>(o_ed, thr + 0, o_sd, o_md, NE);
    sparse_kernel<<<(NE + 255) / 256, 256, 0, stream>>>(o_eu, thr + 1, o_su, o_mu, NE);

    // --- boundary encoding ---
    const int statsN = Bb * R_DIM * H_DIM;
    region_stats_kernel<<<(statsN + 255) / 256, 256, 0, stream>>>(state, rmean, rvar, Bb);
    const int brows = Bb * R_DIM;
    boundary_kernel<<<(brows * BD_DIM + 255) / 256, 256, 0, stream>>>(rmean, rvar, Wm, Wv, o_bd, brows);
}
